// MemristorConv2d_42494406427033
// MI455X (gfx1250) — matrix operand order repaired and checked by reading, not compiled, not run on hardware
//
#include <hip/hip_runtime.h>

typedef _Float16 v16h __attribute__((ext_vector_type(16)));
typedef _Float16 v8h  __attribute__((ext_vector_type(8)));
typedef float    v8f  __attribute__((ext_vector_type(8)));

// Problem constants
#define Bn   16
#define CIN  128
#define COUT 128
#define Fdim 64
#define Tdim 64
#define TP   66              // padded T
#define FP   66              // padded F
#define ASTRIDE 40           // LDS A-tile row stride in halves (80B, bank-conflict-free)
// workspace layout
#define FV_HALFS  ((size_t)Bn * TP * FP * CIN)            // 8,921,088
#define FV_BYTES  (FV_HALFS * 2)                          // 17,842,176 (256B aligned)
#define GW_HALFS  ((size_t)3 * 9 * COUT * CIN)            // 442,368

__device__ __forceinline__ float dac_f(float x) {
    // clip to [-1,1], quantize to 127 levels (RNE like jnp.round), scale by VMAX=0.6
    float c = fminf(fmaxf(x, -1.0f), 1.0f);
    return rintf(c * 127.0f) * (0.6f / 127.0f);
}

__device__ __forceinline__ float adc_f(float y) {
    // clip(round(y*0.01*256)/256, -16, 16)
    return fminf(fmaxf(rintf(y * 2.56f) * 0.00390625f, -16.0f), 16.0f);
}

// ---------------- Kernel 1: zero padded fv workspace (borders) ----------------
__global__ void mem_zero_fv(uint4* __restrict__ p) {
    size_t i = (size_t)blockIdx.x * blockDim.x + threadIdx.x;   // 1,115,136 uint4
    p[i] = make_uint4(0u, 0u, 0u, 0u);
}

// ------- Kernel 2: DAC + cubic, transpose x[B,C,F,T] -> fv[B,T+2,F+2,C] f16 ----
__global__ void mem_fv_fill(const float* __restrict__ x,
                            const float* __restrict__ in_factor,
                            _Float16* __restrict__ fvw) {
    __shared__ _Float16 lds[32][33];
    const int tx = threadIdx.x;           // 0..31
    const int ty = threadIdx.y;           // 0..7
    int tile = blockIdx.x;                // 2 * 4 * 64 * 16 = 8192 tiles
    const int t0 = (tile & 1) * 32;       tile >>= 1;
    const int c0 = (tile & 3) * 32;       tile >>= 2;
    const int f  =  tile & 63;            tile >>= 6;
    const int b  =  tile;

    const float fa = in_factor[0];
#pragma unroll
    for (int j = 0; j < 4; ++j) {
        const int cy = ty + j * 8;
        const float xv = x[(((size_t)b * CIN + (c0 + cy)) * Fdim + f) * Tdim + (t0 + tx)];
        const float v  = dac_f(xv * fa);
        lds[cy][tx] = (_Float16)(v + 0.1f * v * v * v);
    }
    __syncthreads();
#pragma unroll
    for (int j = 0; j < 4; ++j) {
        const int trow = ty + j * 8;      // t offset in tile
        fvw[(((size_t)b * TP + (t0 + trow + 1)) * FP + (f + 1)) * CIN + (c0 + tx)] =
            lds[tx][trow];                // lds[c_off][t_off]
    }
}

// -------- Kernel 3: weights (g_pos-g_neg) -> f16, layout gw[i][tap][o][c] -----
__global__ void mem_w_fill(const float* __restrict__ gp, const float* __restrict__ gn,
                           _Float16* __restrict__ gw) {
    const int idx = blockIdx.x * blockDim.x + threadIdx.x;      // 442,368
    const int c    = idx & 127;
    const int o    = (idx >> 7) & 127;
    const int tapi = idx >> 14;           // 0..26  = i*9 + y*3 + x
    const int i    = tapi / 9;
    const int tap  = tapi - i * 9;
    const int y    = tap / 3;
    const int xx   = tap - y * 3;
    const size_t gin = ((((size_t)i * COUT + o) * CIN + c) * 3 + y) * 3 + xx;
    gw[idx] = (_Float16)(gp[gin] - gn[gin]);
}

// ---------------- Kernel 4: implicit-GEMM conv via WMMA f16 -------------------
// Grid: B * Tp = 1024 blocks, 256 threads (8 waves).
// Block tile: full Fp=64 (M) x full COUT=128 (N), one (b, tp).
// Wave (m2 x n4): 32 f-positions x 32 out-channels x 3 bit-planes = 12 accums.
// A tile (64 rows x 32 ch) triple-buffered in LDS via async global->LDS,
// prefetch distance 2, one barrier per K-step.
__global__ void __launch_bounds__(256)
mem_conv_wmma(const _Float16* __restrict__ fv,
              const _Float16* __restrict__ gw,
              const float* __restrict__ bias,
              const float* __restrict__ out_factor,
              float* __restrict__ out) {
    __shared__ _Float16 abuf[3][64 * ASTRIDE];      // 3 x 5120B

    const int bid = blockIdx.x;
    const int tp  = bid & 63;
    const int b   = bid >> 6;

    const int tid  = threadIdx.x;
    const int w    = tid >> 5;
    const int lane = tid & 31;
    const int m0     = (w & 1) * 32;                // wave's 32 f-rows
    const int n_base = (w >> 1) * 32;               // wave's 32 out-channels

    const int row = lane & 15;                      // A: M row / B,D: N col
    const int hi  = lane >> 4;                      // K-group select per ISA layout

    // cooperative async-load assignment: one b128 (16B) segment per thread
    const int arow = tid >> 2;                      // 0..63  (A tile row)
    const int aseg = tid & 3;                       // 0..3   (8-half segment)

    v8f acc[3][2][2];                               // [plane][n-sub][m-frag]
#pragma unroll
    for (int i = 0; i < 3; ++i)
#pragma unroll
        for (int ns = 0; ns < 2; ++ns)
#pragma unroll
            for (int fg = 0; fg < 2; ++fg)
                acc[i][ns][fg] = (v8f){0.f, 0.f, 0.f, 0.f, 0.f, 0.f, 0.f, 0.f};

    // K-steps: s = tap*4 + cb_idx, tap = y*3+x over 3x3, cb = channel chunk of 32
    // A[r][k] = fv[b][tp+y][r+x][cb+k]   (r = output f position, 0..63)
    auto issue_a = [&](int s, int buf) {
        const int tap = s >> 2;
        const int cb  = (s & 3) << 5;
        const int y   = tap / 3;
        const int xx  = tap - y * 3;
        const _Float16* src =
            fv + (((size_t)b * TP + (tp + y)) * FP + (arow + xx)) * CIN + cb + aseg * 8;
        const unsigned ldsa =
            (unsigned)(uintptr_t)(&abuf[buf][arow * ASTRIDE + aseg * 8]);
        asm volatile("global_load_async_to_lds_b128 %0, %1, off"
                     :: "v"(ldsa), "v"((unsigned long long)(uintptr_t)src)
                     : "memory");
    };

    issue_a(0, 0);                                  // prologue: prefetch depth 2
    issue_a(1, 1);

    for (int s = 0; s < 36; ++s) {
        // wait for this step's tile (own segment); newest prefetch may stay in flight
        if (s < 35) asm volatile("s_wait_asynccnt 0x1" ::: "memory");
        else        asm volatile("s_wait_asynccnt 0x0" ::: "memory");
        __syncthreads();                            // tile[s%3] visible to all waves;
                                                    // also: prior reads of tile[(s+2)%3] done
        if (s + 2 < 36) issue_a(s + 2, (s + 2) % 3);

        const int tap = s >> 2;
        const int cb  = (s & 3) << 5;
        const _Float16* sh = &abuf[s % 3][0];

        // B fragments: 3 planes x 2 n-subtiles, reused across both m-frags.
        // 16-bit B 32x16 layout: lane col N=row, K = hi*16 + 0..15 (contiguous)
        v16h bf[3][2];
#pragma unroll
        for (int i = 0; i < 3; ++i)
#pragma unroll
            for (int ns = 0; ns < 2; ++ns) {
                const int o = n_base + ns * 16 + row;
                bf[i][ns] = *(const v16h*)(
                    gw + (((size_t)(i * 9 + tap) * COUT + o) * CIN) + cb + hi * 16);
            }

#pragma unroll
        for (int fg = 0; fg < 2; ++fg) {
            // 16-bit A 16x32 layout: lane row=lane&15, K = hi*16 + 0..15 (contiguous)
            const int mr = m0 + fg * 16 + row;
            union { v16h v; v8h h[2]; } au;
            au.h[0] = *(const v8h*)(sh + mr * ASTRIDE + hi * 16);
            au.h[1] = *(const v8h*)(sh + mr * ASTRIDE + hi * 16 + 8);
#pragma unroll
            for (int i = 0; i < 3; ++i)
#pragma unroll
                for (int ns = 0; ns < 2; ++ns)
                    acc[i][ns][fg] = __builtin_amdgcn_wmma_f32_16x16x32_f16(
                        false, au.v, false, bf[i][ns], (short)0, acc[i][ns][fg],
                        false, false);
        }
    }

    // Epilogue: ADC each plane, bit-weighted sum, scale, bias, store [B,O,Fp,Tp]
    const float ofac = out_factor[0];
#pragma unroll
    for (int ns = 0; ns < 2; ++ns) {
        const int o  = n_base + ns * 16 + row;
        const float bv = bias[o];
#pragma unroll
        for (int fg = 0; fg < 2; ++fg)
#pragma unroll
            for (int r = 0; r < 8; ++r) {
                const int fp = m0 + fg * 16 + hi * 8 + r;   // D layout: M = r + 8*hi
                const float a0 = adc_f(acc[0][ns][fg][r]);
                const float a1 = adc_f(acc[1][ns][fg][r]);
                const float a2 = adc_f(acc[2][ns][fg][r]);
                const float mem = 4.0f * a0 + 2.0f * a1 + 2.0f * a2;
                out[(((size_t)b * COUT + o) * Fdim + fp) * Tdim + tp] = mem * ofac + bv;
            }
    }
}

extern "C" void kernel_launch(void* const* d_in, const int* in_sizes, int n_in,
                              void* d_out, int out_size, void* d_ws, size_t ws_size,
                              hipStream_t stream) {
    const float* x      = (const float*)d_in[0];
    const float* g_pos  = (const float*)d_in[1];
    const float* g_neg  = (const float*)d_in[2];
    const float* bias   = (const float*)d_in[3];
    const float* in_fac = (const float*)d_in[4];
    const float* out_fac= (const float*)d_in[5];
    float* out = (float*)d_out;

    _Float16* fvw = (_Float16*)d_ws;
    _Float16* gw  = (_Float16*)((char*)d_ws + FV_BYTES);

    // 1) zero padded fv (borders become PAD zeros)
    mem_zero_fv<<<(int)(FV_HALFS / 8 / 256), 256, 0, stream>>>((uint4*)fvw);
    // 2) DAC + cubic + transpose to channels-last padded fv
    mem_fv_fill<<<8192, dim3(32, 8), 0, stream>>>(x, in_fac, fvw);
    // 3) weight prep
    mem_w_fill<<<(int)(GW_HALFS / 256), 256, 0, stream>>>(g_pos, g_neg, gw);
    // 4) WMMA implicit-GEMM conv, all 3 bit-planes fused, async A staging
    mem_conv_wmma<<<Bn * Tdim, 256, 0, stream>>>(fvw, gw, bias, out_fac, out);
}
